// CombinationModel_53154515256115
// MI455X (gfx1250) — compile-verified, matrix operand order checked by reading, not run on hardware
//
#include <hip/hip_runtime.h>
#include <math.h>

// ---------------- problem constants ----------------
#define BB    8
#define TT    300
#define UU    40
#define EE    512            // encoder dim
#define PP    640            // pred dim
#define JJ    640            // joint hidden
#define VV    1056           // vocab
#define DEMB  256
#define UP1   41             // U+1
#define KJ1   1152           // E+P
#define KT1   36             // 1152/32
#define NT1   40             // 640/16
#define KT2   20             // 640/32
#define NT2   66             // 1056/16

#define ROWS_PB   128        // rows per joint block (8 x 16-row tiles)
#define LDH1      (KJ1 + 8)  // LDS row stride (halves), 4-bank rotation
#define LDH2      (JJ  + 8)

typedef __attribute__((ext_vector_type(16))) __bf16 v16bf;
typedef __attribute__((ext_vector_type(8)))  __bf16 v8bf;
typedef __attribute__((ext_vector_type(8)))  float  v8f;

// ---- workspace layout (bytes, all 256B-aligned) ----
#define OFF_W1S   0
#define OFF_W2S   (OFF_W1S + 1474560)       // KJ1*JJ bf16
#define OFF_ENCB  (OFF_W2S + 1351680)       // JJ*VV bf16
#define OFF_PREDB (OFF_ENCB + 2457600)      // B*T*E bf16
#define OFF_H1    (OFF_PREDB + 419840)      // B*UP1*P bf16
#define OFF_H     (OFF_H1 + 839680)         // 328*PP f32 ; then h bf16

// ---------------------------------------------------------------
// prep: bf16 conversion of encoder states + WMMA-fragment swizzle
// of W_j1 / W_j2.  Destination index i decomposes as
//   e = i & 15, lane = (i>>4)&31, t = i>>9, kt = t%KT, nt = t/KT
// ISA 16-bit B layout: n = nt*16 + (lane&15),
//   k = kt*32 + ((lane>>4)<<3) + (e<8 ? e : e+8)
// ---------------------------------------------------------------
__global__ __launch_bounds__(256) void prep_kernel(
    const float* __restrict__ enc, const float* __restrict__ Wj1,
    const float* __restrict__ Wj2, __bf16* __restrict__ encb,
    __bf16* __restrict__ W1s, __bf16* __restrict__ W2s) {
  size_t gid = (size_t)blockIdx.x * blockDim.x + threadIdx.x;
  size_t gsz = (size_t)gridDim.x * blockDim.x;

  for (size_t i = gid; i < (size_t)BB * TT * EE; i += gsz)
    encb[i] = (__bf16)enc[i];

  for (size_t i = gid; i < (size_t)KJ1 * JJ; i += gsz) {
    int e    = (int)(i & 15);
    int lane = (int)((i >> 4) & 31);
    size_t t = i >> 9;
    int kt = (int)(t % KT1), nt = (int)(t / KT1);
    int n = nt * 16 + (lane & 15);
    int k = kt * 32 + ((lane >> 4) << 3) + ((e < 8) ? e : e + 8);
    W1s[i] = (__bf16)Wj1[(size_t)k * JJ + n];
  }

  for (size_t i = gid; i < (size_t)JJ * VV; i += gsz) {
    int e    = (int)(i & 15);
    int lane = (int)((i >> 4) & 31);
    size_t t = i >> 9;
    int kt = (int)(t % KT2), nt = (int)(t / KT2);
    int n = nt * 16 + (lane & 15);
    int k = kt * 32 + ((lane >> 4) << 3) + ((e < 8) ? e : e + 8);
    W2s[i] = (__bf16)Wj2[(size_t)k * VV + n];
  }
}

// ---------------------------------------------------------------
// prediction network (0.3% of FLOPs): plain VALU tanh MLP
// ---------------------------------------------------------------
__global__ __launch_bounds__(256) void pred1_kernel(
    const int* __restrict__ targets, const float* __restrict__ emb,
    const float* __restrict__ W1, const float* __restrict__ b1,
    float* __restrict__ h1) {
  __shared__ float e[2 * DEMB];
  int ro = blockIdx.x;                 // 0..B*UP1-1
  int b = ro / UP1, u = ro % UP1;
  int tid = threadIdx.x;
  int tok0 = (u >= 1) ? targets[b * UU + (u - 1)] : 0;  // ctx i=0
  int tok1 = (u >= 2) ? targets[b * UU + (u - 2)] : 0;  // ctx i=1
  for (int d = tid; d < 2 * DEMB; d += 256) {
    int tok = (d < DEMB) ? tok0 : tok1;
    int dim = d & (DEMB - 1);
    e[d] = tok ? emb[(size_t)tok * DEMB + dim] : 0.0f;   // padding_idx=0
  }
  __syncthreads();
  for (int col = tid; col < PP; col += 256) {
    float acc = b1[col];
    for (int k = 0; k < 2 * DEMB; ++k)
      acc += e[k] * W1[(size_t)k * PP + col];
    h1[(size_t)ro * PP + col] = tanhf(acc);
  }
}

__global__ __launch_bounds__(256) void pred2_kernel(
    const float* __restrict__ h1, const float* __restrict__ W2,
    const float* __restrict__ b2, __bf16* __restrict__ predb) {
  __shared__ float e[PP];
  int ro = blockIdx.x;
  int tid = threadIdx.x;
  for (int d = tid; d < PP; d += 256) e[d] = h1[(size_t)ro * PP + d];
  __syncthreads();
  for (int col = tid; col < PP; col += 256) {
    float acc = b2[col];
    for (int k = 0; k < PP; ++k)
      acc += e[k] * W2[(size_t)k * PP + col];
    predb[(size_t)ro * PP + col] = (__bf16)tanhf(acc);
  }
}

// ---- A fragment from LDS: ISA 16-bit A 16x32 layout ----
__device__ inline v16bf load_a_frag(const __bf16* tile, int lane, int kt, int ldh) {
  int m  = lane & 15;
  int kb = kt * 32 + ((lane >> 4) << 3);
  const __bf16* p = tile + (size_t)m * ldh + kb;
  union { v16bf v; v8bf h[2]; } u;
  u.h[0] = *(const v8bf*)(p);
  u.h[1] = *(const v8bf*)(p + 16);
  return u.v;
}

// ---------------------------------------------------------------
// joint layer 1: gather 128 ragged rows (enc||pred) to LDS once.
// Each wave owns one 16-col tile per pass and accumulates all 8
// row tiles: one B fragment feeds 8 WMMAs (no intra-block B
// redundancy). 5 passes x 36 k-steps. ReLU, bf16 h out.
// ---------------------------------------------------------------
__global__ __launch_bounds__(256) void joint1_kernel(
    const __bf16* __restrict__ encb, const __bf16* __restrict__ predb,
    const __bf16* __restrict__ W1s, const float* __restrict__ bj1,
    __bf16* __restrict__ h, const int* __restrict__ fb,
    const int* __restrict__ ft, const int* __restrict__ fu, int N) {
  __shared__ __align__(16) __bf16 As[ROWS_PB * LDH1];   // 296,960 B
  int tid = threadIdx.x;
  int rowBase = blockIdx.x * ROWS_PB;
  bool full = (rowBase + ROWS_PB) <= N;

  // stage: 1152 halves/row = 144 x 16B chunks
  for (int idx = tid; idx < ROWS_PB * 144; idx += 256) {
    int r = idx / 144, c = idx % 144;
    int g = rowBase + r; if (g >= N) g = N - 1;
    int hoff = c * 8;
    const __bf16* src;
    if (hoff < EE)
      src = encb + ((size_t)fb[g] * TT + ft[g]) * EE + hoff;
    else
      src = predb + ((size_t)fb[g] * UP1 + fu[g]) * PP + (hoff - EE);
    *(uint4*)(&As[(size_t)r * LDH1 + hoff]) = *(const uint4*)src;
  }
  __syncthreads();

  int wave = tid >> 5, lane = tid & 31;
  int n = lane & 15, mb = (lane >> 4) * 8;

  for (int pass = 0; pass < NT1 / 8; ++pass) {
    int nt = pass * 8 + wave;          // each wave: distinct col tile
    int col = nt * 16 + n;
    float bias = bj1[col];
    v8f acc[8];
#pragma unroll
    for (int g = 0; g < 8; ++g)
#pragma unroll
      for (int r = 0; r < 8; ++r) acc[g][r] = bias;

    for (int kt = 0; kt < KT1; ++kt) {
      v16bf bfr = *(const v16bf*)(W1s + (((size_t)nt * KT1 + kt) * 32 + lane) * 16);
#pragma unroll
      for (int g = 0; g < 8; ++g) {
        v16bf a = load_a_frag(&As[(size_t)(g * 16) * LDH1], lane, kt, LDH1);
        acc[g] = __builtin_amdgcn_wmma_f32_16x16x32_bf16(
                     false, a, false, bfr, (short)0, acc[g], false, false);
      }
    }
#pragma unroll
    for (int g = 0; g < 8; ++g) {
      int rbase = rowBase + g * 16 + mb;
      if (full) {
#pragma unroll
        for (int r = 0; r < 8; ++r) {
          float v = acc[g][r];
          if (v < 0.0f) v = 0.0f;      // ReLU
          h[(size_t)(rbase + r) * JJ + col] = (__bf16)v;
        }
      } else {
#pragma unroll
        for (int r = 0; r < 8; ++r) {
          if (rbase + r < N) {
            float v = acc[g][r];
            if (v < 0.0f) v = 0.0f;
            h[(size_t)(rbase + r) * JJ + col] = (__bf16)v;
          }
        }
      }
    }
  }
}

// ---------------------------------------------------------------
// joint layer 2: [N,640] @ [640,1056] + bias -> fp32 out
// 66 col tiles -> 9 passes of 8 waves (wave-uniform tail guard)
// ---------------------------------------------------------------
__global__ __launch_bounds__(256) void joint2_kernel(
    const __bf16* __restrict__ h, const __bf16* __restrict__ W2s,
    const float* __restrict__ bj2, float* __restrict__ out, int N) {
  __shared__ __align__(16) __bf16 As[ROWS_PB * LDH2];   // 165,888 B
  int tid = threadIdx.x;
  int rowBase = blockIdx.x * ROWS_PB;
  bool full = (rowBase + ROWS_PB) <= N;

  for (int idx = tid; idx < ROWS_PB * (JJ / 8); idx += 256) {
    int r = idx / (JJ / 8), c = idx % (JJ / 8);
    int g = rowBase + r; if (g >= N) g = N - 1;
    *(uint4*)(&As[(size_t)r * LDH2 + c * 8]) =
        *(const uint4*)(h + (size_t)g * JJ + c * 8);
  }
  __syncthreads();

  int wave = tid >> 5, lane = tid & 31;
  int n = lane & 15, mb = (lane >> 4) * 8;

  for (int pass = 0; pass < (NT2 + 7) / 8; ++pass) {
    int nt = pass * 8 + wave;
    if (nt >= NT2) continue;           // wave-uniform tail guard
    int col = nt * 16 + n;
    float bias = bj2[col];
    v8f acc[8];
#pragma unroll
    for (int g = 0; g < 8; ++g)
#pragma unroll
      for (int r = 0; r < 8; ++r) acc[g][r] = bias;

    for (int kt = 0; kt < KT2; ++kt) {
      v16bf bfr = *(const v16bf*)(W2s + (((size_t)nt * KT2 + kt) * 32 + lane) * 16);
#pragma unroll
      for (int g = 0; g < 8; ++g) {
        v16bf a = load_a_frag(&As[(size_t)(g * 16) * LDH2], lane, kt, LDH2);
        acc[g] = __builtin_amdgcn_wmma_f32_16x16x32_bf16(
                     false, a, false, bfr, (short)0, acc[g], false, false);
      }
    }
#pragma unroll
    for (int g = 0; g < 8; ++g) {
      int rbase = rowBase + g * 16 + mb;
      if (full) {
#pragma unroll
        for (int r = 0; r < 8; ++r)
          out[(size_t)(rbase + r) * VV + col] = acc[g][r];
      } else {
#pragma unroll
        for (int r = 0; r < 8; ++r)
          if (rbase + r < N)
            out[(size_t)(rbase + r) * VV + col] = acc[g][r];
      }
    }
  }
}

// ---------------------------------------------------------------
extern "C" void kernel_launch(void* const* d_in, const int* in_sizes, int n_in,
                              void* d_out, int out_size, void* d_ws, size_t ws_size,
                              hipStream_t stream) {
  const float* enc     = (const float*)d_in[0];
  const int*   targets = (const int*)d_in[1];
  const float* emb     = (const float*)d_in[2];
  const float* Wp1     = (const float*)d_in[3];
  const float* bp1     = (const float*)d_in[4];
  const float* Wp2     = (const float*)d_in[5];
  const float* bp2     = (const float*)d_in[6];
  const float* Wj1     = (const float*)d_in[7];
  const float* bj1     = (const float*)d_in[8];
  const float* Wj2     = (const float*)d_in[9];
  const float* bj2     = (const float*)d_in[10];
  const int*   fb      = (const int*)d_in[11];
  const int*   ft      = (const int*)d_in[12];
  const int*   fu      = (const int*)d_in[13];
  float* out = (float*)d_out;
  int N = in_sizes[11];                       // 64385 ragged (t,u) pairs

  char* ws = (char*)d_ws;
  __bf16* W1s   = (__bf16*)(ws + OFF_W1S);
  __bf16* W2s   = (__bf16*)(ws + OFF_W2S);
  __bf16* encb  = (__bf16*)(ws + OFF_ENCB);
  __bf16* predb = (__bf16*)(ws + OFF_PREDB);
  float*  h1    = (float*)(ws + OFF_H1);
  __bf16* h     = (__bf16*)(ws + OFF_H);

  prep_kernel<<<2048, 256, 0, stream>>>(enc, Wj1, Wj2, encb, W1s, W2s);
  pred1_kernel<<<BB * UP1, 256, 0, stream>>>(targets, emb, Wp1, bp1, h1);
  pred2_kernel<<<BB * UP1, 256, 0, stream>>>(h1, Wp2, bp2, predb);

  int nRB = (N + ROWS_PB - 1) / ROWS_PB;
  joint1_kernel<<<nRB, 256, 0, stream>>>(encb, predb, W1s, bj1, h, fb, ft, fu, N);
  joint2_kernel<<<nRB, 256, 0, stream>>>(h, W2s, bj2, out, N);
}
